// ami_att_net_82832739270712
// MI455X (gfx1250) — matrix operand order repaired and checked by reading, not compiled, not run on hardware
//
#include <hip/hip_runtime.h>

// ---------------------------------------------------------------------------
// AMI-ATT drug/protein GAT network for MI455X (gfx1250), wave32 + WMMA.
// All dense layers run through one generic f16 WMMA GEMM (v_wmma_f32_16x16x32_f16,
// f32 accumulate), double-buffered in LDS with clamp-based (branch-free) tile
// loads.  GAT segment-softmax is fully deterministic: CSR by dst, per-segment
// lists sorted by edge id, wave-per-node sequential aggregation.
// ---------------------------------------------------------------------------

typedef __attribute__((ext_vector_type(16))) _Float16 v16h;
typedef __attribute__((ext_vector_type(8)))  _Float16 v8h;
typedef __attribute__((ext_vector_type(8)))  float    v8f;

#define B_    64
#define NP_   512
#define NL_   40
#define D_    128
#define ED_   10240
#define EP_   262144
#define NPROT (B_*NP_)   // 32768
#define NDRUG (B_*NL_)   // 2560

static inline int ceilDiv(int a, int b) { return (a + b - 1) / b; }

__device__ __forceinline__ float wave_max(float v) {
    #pragma unroll
    for (int o = 16; o; o >>= 1) v = fmaxf(v, __shfl_xor(v, o, 32));
    return v;
}
__device__ __forceinline__ float wave_sum(float v) {
    #pragma unroll
    for (int o = 16; o; o >>= 1) v += __shfl_xor(v, o, 32);
    return v;
}

// ---------------------------------------------------------------------------
// Generic WMMA GEMM:  out = act( alpha * A[M,K] @ Bt[N,K]^T + bias[N] )
// A, Bt are f16 row-major (K padded to multiple of 32).  Out-of-range rows are
// CLAMPED (not zeroed) on load: an (m,n) output only consumes A row m / Bt row
// n, so garbage only reaches outputs the guarded epilogue never stores.
// Double-buffered LDS (2 x 20KB), one barrier per K-step.
// Optional f32 and f16 outputs with independent relu flags; batched via z.
// ---------------------------------------------------------------------------
__global__ void __launch_bounds__(256)
k_gemm(const _Float16* __restrict__ A, int lda, long long sA,
       const _Float16* __restrict__ Bt, int ldb, long long sB,
       const float* __restrict__ bias,
       float* __restrict__ C, int ldc, long long sC,
       _Float16* __restrict__ Ch, int ldch, long long sCh,
       int M, int N, int K, float alpha, int actC, int actCh)
{
    __shared__ _Float16 As[2][128][40];   // pitch 40 halves = 80B (bank-spread)
    __shared__ _Float16 Bs[2][128][40];

    const int z = blockIdx.z;
    A  += (long long)z * sA;
    Bt += (long long)z * sB;
    float*     Cz  = C  ? C  + (long long)z * sC  : nullptr;
    _Float16*  Chz = Ch ? Ch + (long long)z * sCh : nullptr;

    const int bm = blockIdx.y * 128, bn = blockIdx.x * 128;
    const int tid = threadIdx.x, lane = tid & 31, wid = tid >> 5;
    const int wm = (wid >> 1) * 32, wn = (wid & 1) * 64;
    const int lr = lane & 15, half = lane >> 4;

    // per-thread tile-load coordinates (2 chunks of 8 halves for A and B each)
    int lrow[2], lcol[2];
    const _Float16* aPtr[2];
    const _Float16* bPtr[2];
    #pragma unroll
    for (int ii = 0; ii < 2; ii++) {
        int i = tid + ii * 256;
        lrow[ii] = i >> 2;
        lcol[ii] = (i & 3) * 8;
        int gm = bm + lrow[ii]; if (gm > M - 1) gm = M - 1;   // clamp, branch-free
        int gn = bn + lrow[ii]; if (gn > N - 1) gn = N - 1;
        aPtr[ii] = A  + (size_t)gm * lda + lcol[ii];
        bPtr[ii] = Bt + (size_t)gn * ldb + lcol[ii];
    }

    v8f acc[2][4];
    #pragma unroll
    for (int mi = 0; mi < 2; mi++)
        #pragma unroll
        for (int ni = 0; ni < 4; ni++) { v8f zz = {}; acc[mi][ni] = zz; }

    const int nk = K >> 5;
    v8h va[2], vb[2];
    // prologue: fetch + commit tile 0
    #pragma unroll
    for (int ii = 0; ii < 2; ii++) {
        va[ii] = *(const v8h*)(aPtr[ii]);
        vb[ii] = *(const v8h*)(bPtr[ii]);
    }
    #pragma unroll
    for (int ii = 0; ii < 2; ii++) {
        *(v8h*)(&As[0][lrow[ii]][lcol[ii]]) = va[ii];
        *(v8h*)(&Bs[0][lrow[ii]][lcol[ii]]) = vb[ii];
    }

    int p = 0;
    for (int kt = 0; kt < nk; kt++) {
        __syncthreads();
        const bool more = (kt + 1) < nk;
        if (more) {
            const int k1 = (kt + 1) * 32;
            #pragma unroll
            for (int ii = 0; ii < 2; ii++) {
                va[ii] = *(const v8h*)(aPtr[ii] + k1);
                vb[ii] = *(const v8h*)(bPtr[ii] + k1);
            }
        }

        union Frag { v16h v; v8h h[2]; };
        v16h af[2], bf[4];
        #pragma unroll
        for (int mi = 0; mi < 2; mi++) {
            int r = wm + mi * 16 + lr;
            Frag u;
            u.h[0] = *(const v8h*)(&As[p][r][half * 8]);
            u.h[1] = *(const v8h*)(&As[p][r][16 + half * 8]);
            af[mi] = u.v;
        }
        #pragma unroll
        for (int ni = 0; ni < 4; ni++) {
            int r = wn + ni * 16 + lr;
            Frag u;
            u.h[0] = *(const v8h*)(&Bs[p][r][half * 8]);
            u.h[1] = *(const v8h*)(&Bs[p][r][16 + half * 8]);
            bf[ni] = u.v;
        }
        #pragma unroll
        for (int mi = 0; mi < 2; mi++)
            #pragma unroll
            for (int ni = 0; ni < 4; ni++)
                acc[mi][ni] = __builtin_amdgcn_wmma_f32_16x16x32_f16(
                    false, af[mi], false, bf[ni], (short)0, acc[mi][ni], false, false);

        if (more) {
            int q = p ^ 1;
            #pragma unroll
            for (int ii = 0; ii < 2; ii++) {
                *(v8h*)(&As[q][lrow[ii]][lcol[ii]]) = va[ii];
                *(v8h*)(&Bs[q][lrow[ii]][lcol[ii]]) = vb[ii];
            }
        }
        p ^= 1;
    }

    #pragma unroll
    for (int mi = 0; mi < 2; mi++)
        #pragma unroll
        for (int ni = 0; ni < 4; ni++) {
            int nloc = bn + wn + ni * 16 + lr;
            float bv = 0.f;
            if (bias && nloc < N) bv = bias[nloc];
            #pragma unroll
            for (int i = 0; i < 8; i++) {
                int mloc = bm + wm + mi * 16 + half * 8 + i;
                if (mloc < M && nloc < N) {
                    float v = acc[mi][ni][i] * alpha + bv;
                    if (Cz)  Cz[(size_t)mloc * ldc + nloc]  = actC  ? fmaxf(v, 0.f) : v;
                    if (Chz) Chz[(size_t)mloc * ldch + nloc] = (_Float16)(actCh ? fmaxf(v, 0.f) : v);
                }
            }
        }
}

// --------------------------- utility kernels -------------------------------
__global__ void k_fill_i32(int* p, int v, int n) {
    int i = blockIdx.x * blockDim.x + threadIdx.x;
    if (i < n) p[i] = v;
}
__global__ void k_copy_f32(const float* s, float* d, long long n) {
    for (long long i = (long long)blockIdx.x * blockDim.x + threadIdx.x; i < n;
         i += (long long)gridDim.x * blockDim.x) d[i] = s[i];
}
__global__ void k_cast_f16(const float* s, _Float16* d, long long n) {
    for (long long i = (long long)blockIdx.x * blockDim.x + threadIdx.x; i < n;
         i += (long long)gridDim.x * blockDim.x) d[i] = (_Float16)s[i];
}
// dst[m][0..Kp) = (k<K)? src[m][k] : 0   (f32 -> f16, right-pad K)
__global__ void k_pad_cast(const float* src, int K, _Float16* dst, int Kp, long long total) {
    for (long long i = (long long)blockIdx.x * blockDim.x + threadIdx.x; i < total;
         i += (long long)gridDim.x * blockDim.x) {
        long long m = i / Kp; int k = (int)(i % Kp);
        dst[i] = (k < K) ? (_Float16)src[m * (long long)K + k] : (_Float16)0.f;
    }
}
__global__ void k_copy_pad_f32(const float* s, int n, float* d, int np) {
    int i = blockIdx.x * blockDim.x + threadIdx.x;
    if (i < np) d[i] = (i < n) ? s[i] : 0.f;
}
// copy f32 [rows][cols] (ld sld) into f16 dst at column dcol0 (ld dld)
__global__ void k_cast_cols(const float* s, int sld, _Float16* d, int dld, int dcol0,
                            long long rows, int cols) {
    long long total = rows * (long long)cols;
    for (long long i = (long long)blockIdx.x * blockDim.x + threadIdx.x; i < total;
         i += (long long)gridDim.x * blockDim.x) {
        long long r = i / cols; int c = (int)(i % cols);
        d[r * dld + dcol0 + c] = (_Float16)s[r * sld + c];
    }
}
// tiled transpose with padding: dst[c][r] = (r<R && c<Cc) ? src[r][c] : 0
__global__ void k_transpose_f16(const float* src, int R, int Cc, long long sbs,
                                _Float16* dst, int Rp, int Cp, long long dbs)
{
    __shared__ float tile[32][33];
    src += (long long)blockIdx.z * sbs;
    dst += (long long)blockIdx.z * dbs;
    int r0 = blockIdx.x * 32, c0 = blockIdx.y * 32;
    #pragma unroll
    for (int k = 0; k < 4; k++) {
        int r = r0 + threadIdx.y + 8 * k, c = c0 + threadIdx.x;
        tile[threadIdx.y + 8 * k][threadIdx.x] = (r < R && c < Cc) ? src[(size_t)r * Cc + c] : 0.f;
    }
    __syncthreads();
    #pragma unroll
    for (int k = 0; k < 4; k++) {
        int c = c0 + threadIdx.y + 8 * k, r = r0 + threadIdx.x;
        if (c < Cp && r < Rp) dst[(size_t)c * Rp + r] = (_Float16)tile[threadIdx.x][threadIdx.y + 8 * k];
    }
}

// --------------------------- reductions ------------------------------------
__global__ void k_reduce_part(const float* x, int n, float* part) {
    __shared__ float sh[256];
    float s = 0.f;
    for (int i = blockIdx.x * 256 + threadIdx.x; i < n; i += 256 * 256) s += x[i];
    sh[threadIdx.x] = s; __syncthreads();
    for (int d = 128; d; d >>= 1) { if (threadIdx.x < d) sh[threadIdx.x] += sh[threadIdx.x + d]; __syncthreads(); }
    if (threadIdx.x == 0) part[blockIdx.x] = sh[0];
}
__global__ void k_reduce_fin(const float* part, float inv, float* out) {
    __shared__ float sh[256];
    sh[threadIdx.x] = part[threadIdx.x]; __syncthreads();
    for (int d = 128; d; d >>= 1) { if (threadIdx.x < d) sh[threadIdx.x] += sh[threadIdx.x + d]; __syncthreads(); }
    if (threadIdx.x == 0) out[0] = sh[0] * inv;
}
// c = dot(a, b) over 128 (single wave)
__global__ void k_dot128(const float* a, const float* b, float* out) {
    int lane = threadIdx.x & 31;
    float s = 0.f;
    #pragma unroll
    for (int k = 0; k < 4; k++) s += a[lane + 32 * k] * b[lane + 32 * k];
    s = wave_sum(s);
    if (lane == 0) out[0] = s;
}

// --------------------------- CSR construction ------------------------------
__global__ void k_hist(const int* dst, int E, int* cnt) {
    int i = blockIdx.x * blockDim.x + threadIdx.x;
    if (i < E) atomicAdd(&cnt[dst[i]], 1);
}
__global__ void k_scan(const int* cnt, int n, int* offs) {
    __shared__ int sh[256];
    __shared__ int run;
    if (threadIdx.x == 0) { run = 0; offs[0] = 0; }
    __syncthreads();
    for (int base = 0; base < n; base += 256) {
        int i = base + threadIdx.x;
        int v = (i < n) ? cnt[i] : 0;
        sh[threadIdx.x] = v; __syncthreads();
        for (int d = 1; d < 256; d <<= 1) {
            int t = (threadIdx.x >= d) ? sh[threadIdx.x - d] : 0;
            __syncthreads();
            sh[threadIdx.x] += t;
            __syncthreads();
        }
        if (i < n) offs[i + 1] = run + sh[threadIdx.x];
        __syncthreads();
        if (threadIdx.x == 0) run += sh[255];
        __syncthreads();
    }
}
__global__ void k_scatter(const int* dst, int E, const int* offs, int* cur, int* lists) {
    int i = blockIdx.x * blockDim.x + threadIdx.x;
    if (i < E) {
        int d = dst[i];
        int p = offs[d] + atomicAdd(&cur[d], 1);
        lists[p] = i;
    }
}
__global__ void k_sort_lists(const int* offs, int n, int* lists) {
    int v = blockIdx.x * blockDim.x + threadIdx.x;
    if (v >= n) return;
    int s = offs[v], e = offs[v + 1];
    for (int i = s + 1; i < e; i++) {
        int key = lists[i]; int j = i - 1;
        while (j >= s && lists[j] > key) { lists[j + 1] = lists[j]; j--; }
        lists[j + 1] = key;
    }
}

// --------------------------- GAT kernels -----------------------------------
// a_src[n] = dot(h[n], as), a_dst[n] = dot(h[n], ad)   (wave per node)
__global__ void k_node_coef(const float* h, const float* as_, const float* ad_,
                            float* asrc, float* adst, int n)
{
    int node = blockIdx.x * 8 + (threadIdx.x >> 5);
    int lane = threadIdx.x & 31;
    if (node >= n) return;
    const float* hr = h + (size_t)node * 128;
    float sa = 0.f, sd = 0.f;
    #pragma unroll
    for (int k = 0; k < 4; k++) {
        float v = hr[lane + 32 * k];
        sa += v * as_[lane + 32 * k];
        sd += v * ad_[lane + 32 * k];
    }
    sa = wave_sum(sa); sd = wave_sum(sd);
    if (lane == 0) { asrc[node] = sa; adst[node] = sd; }
}
// e[i] = leaky_relu(a_src[src] + a_dst[dst] + c*ea, 0.2)
__global__ void k_edge_score(const int* src, const int* dst, const float* ea,
                             const float* asrc, const float* adst,
                             const float* cptr, float* e, int E)
{
    int i = blockIdx.x * blockDim.x + threadIdx.x;
    if (i >= E) return;
    float v = asrc[src[i]] + adst[dst[i]] + (*cptr) * ea[i];
    e[i] = v > 0.f ? v : 0.2f * v;
}
// wave-per-node: deterministic segment softmax + weighted h gather (+ self loop)
__global__ void k_gat_agg(const int* offs, const int* lists, const int* src,
                          const float* e, const float* asrc, const float* adst,
                          const float* cptr, const float* meanptr,
                          const float* h, const float* bias,
                          float* accum, _Float16* xout, int n)
{
    int node = blockIdx.x * 8 + (threadIdx.x >> 5);
    int lane = threadIdx.x & 31;
    if (node >= n) return;
    int s = offs[node], t = offs[node + 1];
    float sself = asrc[node] + adst[node] + (*cptr) * (*meanptr);
    sself = sself > 0.f ? sself : 0.2f * sself;
    float m = sself;
    for (int i = s + lane; i < t; i += 32) m = fmaxf(m, e[lists[i]]);
    m = wave_max(m);
    float ssum = (lane == 0) ? __expf(sself - m) : 0.f;
    for (int i = s + lane; i < t; i += 32) ssum += __expf(e[lists[i]] - m);
    ssum = wave_sum(ssum);
    float inv = 1.f / ssum;
    float a0 = 0.f, a1 = 0.f, a2 = 0.f, a3 = 0.f;
    for (int i = s; i < t; i++) {                 // fixed (sorted) order -> deterministic
        int eid = lists[i];
        float w = __expf(e[eid] - m) * inv;
        const float* hr = h + (size_t)src[eid] * 128;
        a0 += w * hr[lane];      a1 += w * hr[lane + 32];
        a2 += w * hr[lane + 64]; a3 += w * hr[lane + 96];
    }
    {
        float w = __expf(sself - m) * inv;
        const float* hr = h + (size_t)node * 128;
        a0 += w * hr[lane];      a1 += w * hr[lane + 32];
        a2 += w * hr[lane + 64]; a3 += w * hr[lane + 96];
    }
    size_t o = (size_t)node * 128;
    #pragma unroll
    for (int k = 0; k < 4; k++) {
        float v = (k == 0 ? a0 : k == 1 ? a1 : k == 2 ? a2 : a3) + bias[lane + 32 * k];
        v = fmaxf(v, 0.f);                        // relu(gat(...))
        xout[o + lane + 32 * k] = (_Float16)v;
        accum[o + lane + 32 * k] += v;
    }
}
__global__ void k_set_col127(const float* dis, float* a, _Float16* ah, int n) {
    int i = blockIdx.x * blockDim.x + threadIdx.x;
    if (i < n) { float v = dis[i]; a[(size_t)i * 128 + 127] = v; ah[(size_t)i * 128 + 127] = (_Float16)v; }
}

// --------------------------- attention / head ------------------------------
__global__ void k_softmax_f16(const float* S, _Float16* P, int rows, int len) {
    int row = blockIdx.x * 8 + (threadIdx.x >> 5);
    int lane = threadIdx.x & 31;
    if (row >= rows) return;
    const float* x = S + (size_t)row * len;
    float m = -3.4e38f;
    for (int i = lane; i < len; i += 32) m = fmaxf(m, x[i]);
    m = wave_max(m);
    float s = 0.f;
    for (int i = lane; i < len; i += 32) s += __expf(x[i] - m);
    s = wave_sum(s);
    float inv = 1.f / s;
    for (int i = lane; i < len; i += 32)
        P[(size_t)row * len + i] = (_Float16)(__expf(x[i] - m) * inv);
}
__global__ void k_layernorm(const float* x, const float* g, const float* b,
                            float* y, int rows)
{
    int row = blockIdx.x * 8 + (threadIdx.x >> 5);
    int lane = threadIdx.x & 31;
    if (row >= rows) return;
    const float* xr = x + (size_t)row * 128;
    float s = 0.f;
    #pragma unroll
    for (int k = 0; k < 4; k++) s += xr[lane + 32 * k];
    float mu = wave_sum(s) * (1.f / 128.f);
    float v = 0.f;
    #pragma unroll
    for (int k = 0; k < 4; k++) { float d = xr[lane + 32 * k] - mu; v += d * d; }
    float var = wave_sum(v) * (1.f / 128.f);
    float rstd = rsqrtf(var + 1e-5f);
    #pragma unroll
    for (int k = 0; k < 4; k++) {
        int c = lane + 32 * k;
        y[(size_t)row * 128 + c] = (xr[c] - mu) * rstd * g[c] + b[c];
    }
}
// xa = x * tanh(x @ fattW + fattb)  (wave per row)
__global__ void k_fatt(const float* x, const float* fw, const float* fb,
                       float* xa, int rows)
{
    int row = blockIdx.x * 8 + (threadIdx.x >> 5);
    int lane = threadIdx.x & 31;
    if (row >= rows) return;
    const float* xr = x + (size_t)row * 128;
    float s = 0.f;
    #pragma unroll
    for (int k = 0; k < 4; k++) s += xr[lane + 32 * k] * fw[lane + 32 * k];
    float t = tanhf(wave_sum(s) + fb[0]);
    #pragma unroll
    for (int k = 0; k < 4; k++) xa[(size_t)row * 128 + lane + 32 * k] = xr[lane + 32 * k] * t;
}
// pooled[b][0:128]=mean over NL, [128:256]=max over NL; also f16 copy
__global__ void k_pool(const float* xa, float* pooled, _Float16* pooledh) {
    int b = blockIdx.x, ch = threadIdx.x;   // 128 threads
    float s = 0.f, mx = -3.4e38f;
    for (int r = 0; r < NL_; r++) {
        float v = xa[((size_t)b * NL_ + r) * 128 + ch];
        s += v; mx = fmaxf(mx, v);
    }
    float mean = s * (1.f / NL_);
    pooled[(size_t)b * 256 + ch] = mean;
    pooled[(size_t)b * 256 + 128 + ch] = mx;
    pooledh[(size_t)b * 256 + ch] = (_Float16)mean;
    pooledh[(size_t)b * 256 + 128 + ch] = (_Float16)mx;
}
// out[row] = dot(x[row], w) + b0  (wave per row, len 128)
__global__ void k_row_dot_out(const float* x, const float* w, const float* b0,
                              float* out, int rows)
{
    int row = blockIdx.x * 8 + (threadIdx.x >> 5);
    int lane = threadIdx.x & 31;
    if (row >= rows) return;
    float s = 0.f;
    #pragma unroll
    for (int k = 0; k < 4; k++) s += x[(size_t)row * 128 + lane + 32 * k] * w[lane + 32 * k];
    s = wave_sum(s);
    if (lane == 0) out[row] = s + b0[0];
}

// ===========================================================================
extern "C" void kernel_launch(void* const* d_in, const int* in_sizes, int n_in,
                              void* d_out, int out_size, void* d_ws, size_t ws_size,
                              hipStream_t stream)
{
    (void)in_sizes; (void)n_in; (void)out_size; (void)ws_size;
    // ---- inputs (setup_inputs dict order; params flattened in insertion order)
    const float* x1   = (const float*)d_in[0];
    const int*   dsrc = (const int*)  d_in[1];
    const int*   ddst = (const int*)  d_in[2];
    const float* dea  = (const float*)d_in[3];
    const float* x3   = (const float*)d_in[4];
    const int*   psrc = (const int*)  d_in[5];
    const int*   pdst = (const int*)  d_in[6];
    const float* pea  = (const float*)d_in[7];
    const float* dis  = (const float*)d_in[8];
    #define PAR(i) ((const float*)d_in[9 + (i)])
    // 0 fc00W 1 fc00b 2 fc01W 3 fc01b 4 fc02W 5 fc02b 6 fc03W 7 fc03b
    // 8 w1 9 b1 10 w2 11 b2 | 12+6j conv(j+1): W,as,ad,We,ae,b
    // 48 wqW 49 wqb 50 wkW 51 wkb 52 wvW 53 wvb 54 caW 55 cab 56 lng 57 lnb
    // 58 fc0W 59 fc0b 60 fattW 61 fattb 62 fc1W 63 fc1b 64 fc2W 65 fc2b
    // 66 fc3W 67 fc3b 68 outW 69 outb
    float* outv = (float*)d_out;

    // ---- workspace carve-up
    char* base = (char*)d_ws;
    size_t off = 0;
    auto alloc = [&](size_t bytes) -> void* {
        off = (off + 255) & ~(size_t)255;
        void* p = base + off; off += bytes; return p;
    };
    _Float16* x3h   = (_Float16*)alloc((size_t)NPROT * 1920 * 2);
    _Float16* Wt00  = (_Float16*)alloc((size_t)1024 * 1920 * 2);
    _Float16* act1h = (_Float16*)alloc((size_t)NPROT * 1024 * 2);
    _Float16* Wt01  = (_Float16*)alloc((size_t)512 * 1024 * 2);
    _Float16* act2h = (_Float16*)alloc((size_t)NPROT * 512 * 2);
    _Float16* Wt02  = (_Float16*)alloc((size_t)128 * 512 * 2);
    float*    b02p  = (float*)   alloc(128 * 4);
    float*    ami0  = (float*)   alloc((size_t)NPROT * 128 * 4);
    _Float16* ami0h = (_Float16*)alloc((size_t)NPROT * 128 * 2);
    float*    accp  = (float*)   alloc((size_t)NPROT * 128 * 4);
    float*    hbuf  = (float*)   alloc((size_t)NPROT * 128 * 4);
    _Float16* xha   = (_Float16*)alloc((size_t)NPROT * 128 * 2);
    _Float16* xhb   = (_Float16*)alloc((size_t)NPROT * 128 * 2);
    float*    asrc  = (float*)   alloc((size_t)NPROT * 4);
    float*    adst  = (float*)   alloc((size_t)NPROT * 4);
    float*    escr  = (float*)   alloc((size_t)EP_ * 4);
    int* pCnt  = (int*)alloc((size_t)NPROT * 4);
    int* pCur  = (int*)alloc((size_t)NPROT * 4);
    int* pOffs = (int*)alloc((size_t)(NPROT + 1) * 4);
    int* pList = (int*)alloc((size_t)EP_ * 4);
    int* dCnt  = (int*)alloc((size_t)NDRUG * 4);
    int* dCur  = (int*)alloc((size_t)NDRUG * 4);
    int* dOffs = (int*)alloc((size_t)(NDRUG + 1) * 4);
    int* dList = (int*)alloc((size_t)ED_ * 4);
    float* part  = (float*)alloc(256 * 4);
    float* means = (float*)alloc(2 * 4);      // [0]=drug mean(ea), [1]=prot
    float* cvals = (float*)alloc(6 * 4);      // dot(We, ae) per conv1..6
    _Float16* Wtconv[6];
    for (int j = 0; j < 6; j++) Wtconv[j] = (_Float16*)alloc((size_t)128 * 128 * 2);
    _Float16* x1h    = (_Float16*)alloc((size_t)NDRUG * 64 * 2);
    _Float16* Wtw1   = (_Float16*)alloc((size_t)128 * 64 * 2);
    _Float16* h1h    = (_Float16*)alloc((size_t)NDRUG * 128 * 2);
    _Float16* Wtw2   = (_Float16*)alloc((size_t)64 * 128 * 2);
    _Float16* h2hh   = (_Float16*)alloc((size_t)NDRUG * 64 * 2);
    _Float16* Wtf03  = (_Float16*)alloc((size_t)128 * 64 * 2);
    float*    x10    = (float*)   alloc((size_t)NDRUG * 128 * 4);
    _Float16* x10h   = (_Float16*)alloc((size_t)NDRUG * 128 * 2);
    float*    accd   = (float*)   alloc((size_t)NDRUG * 128 * 4);
    _Float16* dxa    = (_Float16*)alloc((size_t)NDRUG * 128 * 2);
    _Float16* dxb    = (_Float16*)alloc((size_t)NDRUG * 128 * 2);
    _Float16* dsumh  = (_Float16*)alloc((size_t)NDRUG * 128 * 2);
    _Float16* psumh  = (_Float16*)alloc((size_t)NPROT * 128 * 2);
    _Float16* WtQ    = (_Float16*)alloc((size_t)128 * 128 * 2);
    _Float16* WtK    = (_Float16*)alloc((size_t)128 * 128 * 2);
    _Float16* WtV    = (_Float16*)alloc((size_t)128 * 128 * 2);
    _Float16* WtCA   = (_Float16*)alloc((size_t)128 * 128 * 2);
    _Float16* Qh     = (_Float16*)alloc((size_t)NDRUG * 128 * 2);
    _Float16* Kh     = (_Float16*)alloc((size_t)NPROT * 128 * 2);
    float*    Vf     = (float*)   alloc((size_t)NPROT * 128 * 4);
    _Float16* Vt     = (_Float16*)alloc((size_t)B_ * 128 * 512 * 2);
    float*    Smat   = (float*)   alloc((size_t)B_ * NL_ * 512 * 4);
    _Float16* attnh  = (_Float16*)alloc((size_t)B_ * NL_ * 512 * 2);
    _Float16* Ph     = (_Float16*)alloc((size_t)NDRUG * 128 * 2);
    float*    attf   = (float*)   alloc((size_t)NDRUG * 128 * 4);
    float*    amiatt = (float*)   alloc((size_t)NDRUG * 128 * 4);
    _Float16* Wtfc0  = (_Float16*)alloc((size_t)128 * 256 * 2);
    _Float16* cath   = (_Float16*)alloc((size_t)NDRUG * 256 * 2);
    float*    x0f    = (float*)   alloc((size_t)NDRUG * 128 * 4);
    float*    xaf    = (float*)   alloc((size_t)NDRUG * 128 * 4);
    float*    pooled = (float*)   alloc((size_t)B_ * 256 * 4);
    _Float16* poolh  = (_Float16*)alloc((size_t)B_ * 256 * 2);
    _Float16* Wtfc1  = (_Float16*)alloc((size_t)512 * 256 * 2);
    _Float16* h2rh   = (_Float16*)alloc((size_t)B_ * 512 * 2);
    _Float16* Wtfc2  = (_Float16*)alloc((size_t)256 * 512 * 2);
    _Float16* x2h    = (_Float16*)alloc((size_t)B_ * 256 * 2);
    _Float16* Wtfc3  = (_Float16*)alloc((size_t)128 * 256 * 2);
    float*    x3f    = (float*)   alloc((size_t)B_ * 128 * 4);

    auto gemm = [&](const _Float16* A, int lda, long long sA,
                    const _Float16* Bt, int ldb, long long sB,
                    const float* bias,
                    float* C, int ldc, long long sC,
                    _Float16* Ch, int ldch, long long sCh,
                    int M, int N, int K, float alpha, int actC, int actCh, int batch) {
        dim3 g(ceilDiv(N, 128), ceilDiv(M, 128), batch);
        k_gemm<<<g, 256, 0, stream>>>(A, lda, sA, Bt, ldb, sB, bias,
                                      C, ldc, sC, Ch, ldch, sCh, M, N, K, alpha, actC, actCh);
    };
    auto transpose = [&](const float* src, int R, int Cc, long long sbs,
                         _Float16* dst, int Rp, int Cp, long long dbs, int batch) {
        dim3 g(ceilDiv(Rp, 32), ceilDiv(Cp, 32), batch);
        k_transpose_f16<<<g, dim3(32, 8), 0, stream>>>(src, R, Cc, sbs, dst, Rp, Cp, dbs);
    };

    // ================= CSR construction (deterministic) =================
    k_fill_i32<<<ceilDiv(NPROT, 256), 256, 0, stream>>>(pCnt, 0, NPROT);
    k_fill_i32<<<ceilDiv(NPROT, 256), 256, 0, stream>>>(pCur, 0, NPROT);
    k_fill_i32<<<ceilDiv(NDRUG, 256), 256, 0, stream>>>(dCnt, 0, NDRUG);
    k_fill_i32<<<ceilDiv(NDRUG, 256), 256, 0, stream>>>(dCur, 0, NDRUG);
    k_hist<<<ceilDiv(EP_, 256), 256, 0, stream>>>(pdst, EP_, pCnt);
    k_hist<<<ceilDiv(ED_, 256), 256, 0, stream>>>(ddst, ED_, dCnt);
    k_scan<<<1, 256, 0, stream>>>(pCnt, NPROT, pOffs);
    k_scan<<<1, 256, 0, stream>>>(dCnt, NDRUG, dOffs);
    k_scatter<<<ceilDiv(EP_, 256), 256, 0, stream>>>(pdst, EP_, pOffs, pCur, pList);
    k_scatter<<<ceilDiv(ED_, 256), 256, 0, stream>>>(ddst, ED_, dOffs, dCur, dList);
    k_sort_lists<<<ceilDiv(NPROT, 256), 256, 0, stream>>>(pOffs, NPROT, pList);
    k_sort_lists<<<ceilDiv(NDRUG, 256), 256, 0, stream>>>(dOffs, NDRUG, dList);

    // edge-attr means + per-conv c = dot(We, ae)
    k_reduce_part<<<256, 256, 0, stream>>>(dea, ED_, part);
    k_reduce_fin<<<1, 256, 0, stream>>>(part, 1.f / ED_, &means[0]);
    k_reduce_part<<<256, 256, 0, stream>>>(pea, EP_, part);
    k_reduce_fin<<<1, 256, 0, stream>>>(part, 1.f / EP_, &means[1]);
    for (int j = 0; j < 6; j++)
        k_dot128<<<1, 32, 0, stream>>>(PAR(12 + 6 * j + 3), PAR(12 + 6 * j + 4), &cvals[j]);

    // ================= weight conversion / transposition =================
    transpose(PAR(0), 1900, 1024, 0, Wt00, 1920, 1024, 0, 1);   // fc00W
    transpose(PAR(2), 1024, 512, 0, Wt01, 1024, 512, 0, 1);     // fc01W
    transpose(PAR(4), 512, 127, 0, Wt02, 512, 128, 0, 1);       // fc02W (pad N->128)
    k_copy_pad_f32<<<1, 128, 0, stream>>>(PAR(5), 127, b02p, 128);
    transpose(PAR(8), 52, 128, 0, Wtw1, 64, 128, 0, 1);         // w1 (pad K->64)
    transpose(PAR(10), 128, 64, 0, Wtw2, 128, 64, 0, 1);        // w2
    transpose(PAR(6), 64, 128, 0, Wtf03, 64, 128, 0, 1);        // fc03W
    for (int j = 0; j < 6; j++)
        transpose(PAR(12 + 6 * j), 128, 128, 0, Wtconv[j], 128, 128, 0, 1);
    transpose(PAR(48), 128, 128, 0, WtQ, 128, 128, 0, 1);
    transpose(PAR(50), 128, 128, 0, WtK, 128, 128, 0, 1);
    transpose(PAR(52), 128, 128, 0, WtV, 128, 128, 0, 1);
    transpose(PAR(54), 128, 128, 0, WtCA, 128, 128, 0, 1);
    transpose(PAR(58), 256, 128, 0, Wtfc0, 256, 128, 0, 1);
    transpose(PAR(62), 256, 512, 0, Wtfc1, 256, 512, 0, 1);
    transpose(PAR(64), 512, 256, 0, Wtfc2, 512, 256, 0, 1);
    transpose(PAR(66), 256, 128, 0, Wtfc3, 256, 128, 0, 1);

    // ================= protein branch =================
    k_pad_cast<<<65535, 256, 0, stream>>>(x3, 1900, x3h, 1920, (long long)NPROT * 1920);
    gemm(x3h, 1920, 0, Wt00, 1920, 0, PAR(1), nullptr, 0, 0, act1h, 1024, 0,
         NPROT, 1024, 1920, 1.f, 0, 1, 1);                      // relu
    gemm(act1h, 1024, 0, Wt01, 1024, 0, PAR(3), nullptr, 0, 0, act2h, 512, 0,
         NPROT, 512, 1024, 1.f, 0, 1, 1);                       // relu
    gemm(act2h, 512, 0, Wt02, 512, 0, b02p, ami0, 128, 0, ami0h, 128, 0,
         NPROT, 128, 512, 1.f, 1, 1, 1);                        // relu, col 127 = 0
    k_set_col127<<<ceilDiv(NPROT, 256), 256, 0, stream>>>(dis, ami0, ami0h, NPROT);
    k_copy_f32<<<16384, 256, 0, stream>>>(ami0, accp, (long long)NPROT * 128);

    {   // conv4..conv6
        const _Float16* xin = ami0h;
        _Float16* xouts[3] = { xha, xhb, xha };
        for (int l = 0; l < 3; l++) {
            int cj = 3 + l;                                     // conv index 0-based
            gemm(xin, 128, 0, Wtconv[cj], 128, 0, nullptr, hbuf, 128, 0,
                 nullptr, 0, 0, NPROT, 128, 128, 1.f, 0, 0, 1);
            k_node_coef<<<ceilDiv(NPROT, 8), 256, 0, stream>>>(
                hbuf, PAR(12 + 6 * cj + 1), PAR(12 + 6 * cj + 2), asrc, adst, NPROT);
            k_edge_score<<<ceilDiv(EP_, 256), 256, 0, stream>>>(
                psrc, pdst, pea, asrc, adst, &cvals[cj], escr, EP_);
            k_gat_agg<<<ceilDiv(NPROT, 8), 256, 0, stream>>>(
                pOffs, pList, psrc, escr, asrc, adst, &cvals[cj], &means[1],
                hbuf, PAR(12 + 6 * cj + 5), accp, xouts[l], NPROT);
            xin = xouts[l];
        }
    }

    // ================= drug branch =================
    k_pad_cast<<<ceilDiv(NDRUG * 64, 256), 256, 0, stream>>>(x1, 52, x1h, 64, (long long)NDRUG * 64);
    gemm(x1h, 64, 0, Wtw1, 64, 0, PAR(9), nullptr, 0, 0, h1h, 128, 0,
         NDRUG, 128, 64, 1.f, 0, 1, 1);                         // relu
    gemm(h1h, 128, 0, Wtw2, 128, 0, PAR(11), nullptr, 0, 0, h2hh, 64, 0,
         NDRUG, 64, 128, 1.f, 0, 1, 1);                         // relu
    gemm(h2hh, 64, 0, Wtf03, 64, 0, PAR(7), x10, 128, 0, x10h, 128, 0,
         NDRUG, 128, 64, 1.f, 0, 0, 1);                         // no act
    k_copy_f32<<<1280, 256, 0, stream>>>(x10, accd, (long long)NDRUG * 128);

    {   // conv1..conv3
        const _Float16* xin = x10h;
        _Float16* xouts[3] = { dxa, dxb, dxa };
        for (int l = 0; l < 3; l++) {
            int cj = l;
            gemm(xin, 128, 0, Wtconv[cj], 128, 0, nullptr, hbuf, 128, 0,
                 nullptr, 0, 0, NDRUG, 128, 128, 1.f, 0, 0, 1);
            k_node_coef<<<ceilDiv(NDRUG, 8), 256, 0, stream>>>(
                hbuf, PAR(12 + 6 * cj + 1), PAR(12 + 6 * cj + 2), asrc, adst, NDRUG);
            k_edge_score<<<ceilDiv(ED_, 256), 256, 0, stream>>>(
                dsrc, ddst, dea, asrc, adst, &cvals[cj], escr, ED_);
            k_gat_agg<<<ceilDiv(NDRUG, 8), 256, 0, stream>>>(
                dOffs, dList, dsrc, escr, asrc, adst, &cvals[cj], &means[0],
                hbuf, PAR(12 + 6 * cj + 5), accd, xouts[l], NDRUG);
            xin = xouts[l];
        }
    }

    // ================= cross attention =================
    k_cast_f16<<<1280, 256, 0, stream>>>(accd, dsumh, (long long)NDRUG * 128);
    k_cast_f16<<<16384, 256, 0, stream>>>(accp, psumh, (long long)NPROT * 128);
    gemm(dsumh, 128, 0, WtQ, 128, 0, PAR(49), nullptr, 0, 0, Qh, 128, 0,
         NDRUG, 128, 128, 1.f, 0, 0, 1);
    gemm(psumh, 128, 0, WtK, 128, 0, PAR(51), nullptr, 0, 0, Kh, 128, 0,
         NPROT, 128, 128, 1.f, 0, 0, 1);
    gemm(psumh, 128, 0, WtV, 128, 0, PAR(53), Vf, 128, 0, nullptr, 0, 0,
         NPROT, 128, 128, 1.f, 0, 0, 1);
    transpose(Vf, 512, 128, 512LL * 128, Vt, 512, 128, 128LL * 512, B_);  // V^T per batch
    // S = Q K^T / sqrt(D)  (batched; Kh rows are already [N][K] layout)
    gemm(Qh, 128, (long long)NL_ * 128, Kh, 128, 512LL * 128, nullptr,
         Smat, 512, (long long)NL_ * 512, nullptr, 0, 0,
         NL_, 512, 128, 0.08838834764831845f, 0, 0, B_);
    k_softmax_f16<<<ceilDiv(NDRUG, 8), 256, 0, stream>>>(Smat, attnh, NDRUG, 512);
    // P = attn @ V  (batched)
    gemm(attnh, 512, (long long)NL_ * 512, Vt, 512, 128LL * 512, nullptr,
         nullptr, 0, 0, Ph, 128, (long long)NL_ * 128,
         NL_, 128, 512, 1.f, 0, 0, B_);
    gemm(Ph, 128, 0, WtCA, 128, 0, PAR(55), attf, 128, 0, nullptr, 0, 0,
         NDRUG, 128, 128, 1.f, 0, 0, 1);
    k_layernorm<<<ceilDiv(NDRUG, 8), 256, 0, stream>>>(attf, PAR(56), PAR(57), amiatt, NDRUG);

    // ================= pooling head =================
    k_cast_cols<<<1280, 256, 0, stream>>>(accd, 128, cath, 256, 0,   NDRUG, 128);
    k_cast_cols<<<1280, 256, 0, stream>>>(amiatt, 128, cath, 256, 128, NDRUG, 128);
    gemm(cath, 256, 0, Wtfc0, 256, 0, PAR(59), x0f, 128, 0, nullptr, 0, 0,
         NDRUG, 128, 256, 1.f, 1, 0, 1);                        // relu
    k_fatt<<<ceilDiv(NDRUG, 8), 256, 0, stream>>>(x0f, PAR(60), PAR(61), xaf, NDRUG);
    k_pool<<<B_, 128, 0, stream>>>(xaf, pooled, poolh);
    // h2 (pre-relu) goes straight into d_out[64:]; relu(h2) f16 for fc2
    gemm(poolh, 256, 0, Wtfc1, 256, 0, PAR(63), outv + 64, 512, 0, h2rh, 512, 0,
         B_, 512, 256, 1.f, 0, 1, 1);
    gemm(h2rh, 512, 0, Wtfc2, 512, 0, PAR(65), nullptr, 0, 0, x2h, 256, 0,
         B_, 256, 512, 1.f, 0, 1, 1);                           // relu
    gemm(x2h, 256, 0, Wtfc3, 256, 0, PAR(67), x3f, 128, 0, nullptr, 0, 0,
         B_, 128, 256, 1.f, 1, 0, 1);                           // relu
    k_row_dot_out<<<ceilDiv(B_, 8), 256, 0, stream>>>(x3f, PAR(68), PAR(69), outv, B_);
    #undef PAR
}
